// TemporalCrossTransformer_48163763258190
// MI455X (gfx1250) — matrix operand order repaired and checked by reading, not compiled, not run on hardware
//
#include <hip/hip_runtime.h>
#include <hip/hip_bf16.h>

typedef __attribute__((ext_vector_type(16))) __bf16 v16bf;
typedef __attribute__((ext_vector_type(8)))  float  v8f;

#define B_    16
#define NS_   25
#define NQ_   20
#define SEQ   8
#define IND   1024
#define OUTD  1152
#define KDIM  2048
#define TT    28   // number of tuples C(8,2)

__constant__ int dTup0[28] = {0,0,0,0,0,0,0,1,1,1,1,1,1,2,2,2,2,2,3,3,3,3,4,4,4,5,5,6};
__constant__ int dTup1[28] = {1,2,3,4,5,6,7,2,3,4,5,6,7,3,4,5,6,7,4,5,6,7,5,6,7,6,7,7};

__device__ __forceinline__ unsigned short f2bf(float f) {
    unsigned int u = __float_as_uint(f);
    u += 0x7FFFu + ((u >> 16) & 1u);
    return (unsigned short)(u >> 16);
}

union BFrag { v16bf v; float4 f[2]; };

__device__ __forceinline__ v8f wmma_bf16(v16bf a, v16bf b, v8f c) {
    return __builtin_amdgcn_wmma_f32_16x16x32_bf16(false, a, false, b, (short)0, c, false, false);
}

// ---------------- prep: x + positional encoding -> bf16 -------------------
__global__ void addpe_bf16_kernel(const float* __restrict__ src,
                                  unsigned short* __restrict__ dst, int total) {
    int idx = blockIdx.x * blockDim.x + threadIdx.x;
    if (idx >= total) return;
    int d = idx & (IND - 1);
    int s = (idx >> 10) & (SEQ - 1);
    int i2 = d & ~1;
    // ln(10000)/1024 = 0.0089944730f
    float div = __expf(-(float)i2 * 0.0089944730f);
    float ang = (float)s * div;
    float pe  = ((d & 1) ? __cosf(ang) : __sinf(ang)) * 0.1f;
    dst[idx] = f2bf(src[idx] + pe);
}

// --------------- prep: weight (2048x1152) -> N-major bf16 (1152x2048) -----
__global__ void wt_transpose_kernel(const float* __restrict__ w,
                                    unsigned short* __restrict__ wt, int total) {
    int idx = blockIdx.x * blockDim.x + threadIdx.x;
    if (idx >= total) return;
    int n = idx / KDIM;
    int k = idx - n * KDIM;
    wt[idx] = f2bf(w[(size_t)k * OUTD + n]);
}

// --------------- projection GEMM (+optional fused layernorm) --------------
// grid.x = B * nItems * 2 (m-tile), block = 256 (8 waves, 9 n-tiles each)
__launch_bounds__(256)
__global__ void proj_kernel(const unsigned short* __restrict__ X,   // (B,nItems,8,1024) bf16
                            const unsigned short* __restrict__ Wt,  // (1152,2048) bf16 N-major
                            const float* __restrict__ bias,
                            const float* __restrict__ g,
                            const float* __restrict__ bb,
                            unsigned short* __restrict__ outB,      // (B,nItems,28,1152) bf16 (may be null)
                            float* __restrict__ outF,               // same shape f32 (may be null)
                            int nItems, int doLN)
{
    __shared__ float sSum[16];
    __shared__ float sSq[16];

    int blk   = blockIdx.x;
    int mtile = blk & 1;
    int item  = (blk >> 1) % nItems;
    int b     = (blk >> 1) / nItems;
    int tid   = threadIdx.x;
    int lane  = tid & 31;
    int wv    = tid >> 5;          // wave 0..7
    int half  = lane >> 4;
    int r     = lane & 15;

    int mg = mtile * 16 + r;
    int mc = mg < TT ? mg : TT - 1;
    int s0 = dTup0[mc], s1 = dTup1[mc];
    const unsigned short* Xb    = X + (size_t)(b * nItems + item) * SEQ * IND;
    const unsigned short* rowLo = Xb + (size_t)s0 * IND;
    const unsigned short* rowHi = Xb + (size_t)s1 * IND;

    v8f acc[9];
    v8f z8 = {0.f,0.f,0.f,0.f,0.f,0.f,0.f,0.f};
#pragma unroll
    for (int j = 0; j < 9; ++j) acc[j] = z8;

    int nbase = wv * 144;          // 9 tiles * 16

#pragma unroll 1
    for (int ks = 0; ks < 64; ++ks) {
        int k0 = ks * 32;
        const unsigned short* arow = (k0 < IND) ? (rowLo + k0) : (rowHi + (k0 - IND));
        BFrag a;
        a.f[0] = *(const float4*)(const void*)(arow + half * 16);
        a.f[1] = *(const float4*)(const void*)(arow + half * 16 + 8);
#pragma unroll
        for (int j = 0; j < 9; ++j) {
            const unsigned short* brow =
                Wt + (size_t)(nbase + j * 16 + r) * KDIM + k0 + half * 16;
            BFrag bf;
            bf.f[0] = *(const float4*)(const void*)(brow);
            bf.f[1] = *(const float4*)(const void*)(brow + 8);
            acc[j] = wmma_bf16(a.v, bf.v, acc[j]);
        }
    }

    // add bias
#pragma unroll
    for (int j = 0; j < 9; ++j) {
        float bv = bias[nbase + j * 16 + r];
#pragma unroll
        for (int rr = 0; rr < 8; ++rr) acc[j][rr] += bv;
    }

    if (doLN) {
        if (tid < 16) { sSum[tid] = 0.f; sSq[tid] = 0.f; }
        __syncthreads();
#pragma unroll
        for (int rr = 0; rr < 8; ++rr) {
            int m = half * 8 + rr;
            float p = 0.f, p2 = 0.f;
#pragma unroll
            for (int j = 0; j < 9; ++j) { float v = acc[j][rr]; p += v; p2 += v * v; }
            atomicAdd(&sSum[m], p);
            atomicAdd(&sSq[m], p2);
        }
        __syncthreads();
    }

    size_t rowBase = (size_t)(b * nItems + item) * TT;
#pragma unroll
    for (int rr = 0; rr < 8; ++rr) {
        int m   = half * 8 + rr;
        int mgl = mtile * 16 + m;
        if (mgl >= TT) continue;
        float mean = 0.f, rstd = 1.f;
        if (doLN) {
            mean = sSum[m] * (1.0f / OUTD);
            float var = sSq[m] * (1.0f / OUTD) - mean * mean;
            rstd = rsqrtf(var + 1e-5f);
        }
#pragma unroll
        for (int j = 0; j < 9; ++j) {
            int n = nbase + j * 16 + r;
            float v = acc[j][rr];
            if (doLN) v = (v - mean) * rstd * g[n] + bb[n];
            size_t off = (rowBase + mgl) * OUTD + n;
            if (outB) outB[off] = f2bf(v);
            if (outF) outF[off] = v;
        }
    }
}

// --------- transpose s_v (b,n,28,1152) -> (b,n,1152,32) zero-padded -------
__global__ void sv_transpose_kernel(const unsigned short* __restrict__ sv,
                                    unsigned short* __restrict__ svt) {
    int bn = blockIdx.x;
    const unsigned short* src = sv + (size_t)bn * TT * OUTD;
    unsigned short* dst = svt + (size_t)bn * OUTD * 32;
    for (int idx = threadIdx.x; idx < OUTD * 32; idx += blockDim.x) {
        int O = idx >> 5;
        int s = idx & 31;
        dst[idx] = (s < TT) ? src[(size_t)s * OUTD + O] : (unsigned short)0;
    }
}

// --------- attention: block = 4 waves sharing one (b,c); wave w -> q ------
__launch_bounds__(128)
__global__ void attn_kernel(const unsigned short* __restrict__ qk,   // (B,NQ,28,1152) bf16
                            const unsigned short* __restrict__ sk,   // (B,NS,28,1152) bf16
                            const unsigned short* __restrict__ svt,  // (B,NS,1152,32) bf16
                            const float* __restrict__ qv,            // (B,NQ,28,1152) f32
                            float* __restrict__ logits,
                            float* __restrict__ diffs,
                            float* __restrict__ protos)
{
    __shared__ __align__(16) unsigned short sV[OUTD * 32];  // class_v^T tile, 73728 B
    __shared__ __align__(16) float          sS[4][1024];
    __shared__ __align__(16) unsigned short sB[4][1024];

    int tid  = threadIdx.x;
    int w    = tid >> 5;
    int lane = tid & 31;
    // block -> (b, c, qg); wave w -> q = qg*4 + w
    int qg   = blockIdx.x % (NQ_ / 4);
    int c    = (blockIdx.x / (NQ_ / 4)) % NS_;
    int b    = blockIdx.x / ((NQ_ / 4) * NS_);
    int q    = qg * 4 + w;
    int half = lane >> 4;
    int r    = lane & 15;

    float* S = sS[w];
    unsigned short* Bsm = sB[w];

    const unsigned short* A  = qk + (size_t)(b * NQ_ + q) * TT * OUTD;
    const unsigned short* Bk = sk + (size_t)(b * NS_ + c) * TT * OUTD;
    const unsigned short* Bv = svt + (size_t)(b * NS_ + c) * OUTD * 32;

    // ---- async stage of class_v^T tile into LDS (ASYNCcnt path) ----
    {
        unsigned ldsbase = (unsigned)(unsigned long long)(&sV[0]);
        // 73728 bytes = 4608 x 16B chunks; 128 threads -> 36 chunks each
#pragma unroll 1
        for (int it = 0; it < 36; ++it) {
            int ci = tid + it * 128;
            unsigned long long ga = (unsigned long long)(const void*)(Bv + ci * 8);
            unsigned la = ldsbase + (unsigned)(ci * 16);
            asm volatile("global_load_async_to_lds_b128 %0, %1, off"
                         :: "v"(la), "v"(ga) : "memory");
        }
        asm volatile("s_wait_asynccnt 0x0" ::: "memory");
    }

    // ---- phase 1: scores[t][s] = q_k[t,:] . class_k[s,:] ----
    v8f z8 = {0.f,0.f,0.f,0.f,0.f,0.f,0.f,0.f};
    v8f acc[2][2];
    acc[0][0] = z8; acc[0][1] = z8; acc[1][0] = z8; acc[1][1] = z8;

#pragma unroll 1
    for (int ks = 0; ks < 36; ++ks) {
        int k0 = ks * 32;
        BFrag a0, a1, b0, b1;
        const unsigned short* ar0 = A  + (size_t)(r)      * OUTD + k0;
        const unsigned short* ar1 = A  + (size_t)(16 + r) * OUTD + k0;
        const unsigned short* br0 = Bk + (size_t)(r)      * OUTD + k0;
        const unsigned short* br1 = Bk + (size_t)(16 + r) * OUTD + k0;
        a0.f[0] = *(const float4*)(const void*)(ar0 + half * 16);
        a0.f[1] = *(const float4*)(const void*)(ar0 + half * 16 + 8);
        a1.f[0] = *(const float4*)(const void*)(ar1 + half * 16);
        a1.f[1] = *(const float4*)(const void*)(ar1 + half * 16 + 8);
        b0.f[0] = *(const float4*)(const void*)(br0 + half * 16);
        b0.f[1] = *(const float4*)(const void*)(br0 + half * 16 + 8);
        b1.f[0] = *(const float4*)(const void*)(br1 + half * 16);
        b1.f[1] = *(const float4*)(const void*)(br1 + half * 16 + 8);
        acc[0][0] = wmma_bf16(a0.v, b0.v, acc[0][0]);
        acc[0][1] = wmma_bf16(a0.v, b1.v, acc[0][1]);
        acc[1][0] = wmma_bf16(a1.v, b0.v, acc[1][0]);
        acc[1][1] = wmma_bf16(a1.v, b1.v, acc[1][1]);
    }

    const float scale = rsqrtf((float)OUTD);
#pragma unroll
    for (int mi = 0; mi < 2; ++mi)
#pragma unroll
        for (int sj = 0; sj < 2; ++sj)
#pragma unroll
            for (int rr = 0; rr < 8; ++rr) {
                int t = mi * 16 + half * 8 + rr;
                int s = sj * 16 + r;
                S[t * 32 + s] = acc[mi][sj][rr] * scale;
            }
    __syncthreads();   // also makes async-staged sV visible to all waves

    // ---- phase 2: softmax over t (axis=-2) per column s ----
    {
        int s = lane;
        if (s < TT) {
            float x[TT];
            float mx = -1e30f;
#pragma unroll
            for (int t = 0; t < TT; ++t) { x[t] = S[t * 32 + s]; mx = fmaxf(mx, x[t]); }
            float sum = 0.f;
#pragma unroll
            for (int t = 0; t < TT; ++t) { x[t] = __expf(x[t] - mx); sum += x[t]; }
            float inv = 1.0f / sum;
#pragma unroll
            for (int t = 0; t < TT; ++t) Bsm[t * 32 + s] = f2bf(x[t] * inv);
#pragma unroll
            for (int t = TT; t < 32; ++t) Bsm[t * 32 + s] = 0;
        } else {
#pragma unroll
            for (int t = 0; t < 32; ++t) Bsm[t * 32 + s] = 0;
        }
    }
    __syncthreads();

    // ---- phase 3: proto[t,O] = sum_s scores[t,s] * class_v[s,O] ----
    BFrag pa[2];
#pragma unroll
    for (int mi = 0; mi < 2; ++mi) {
        const unsigned short* prow = &Bsm[(mi * 16 + r) * 32];
        pa[mi].f[0] = *(const float4*)(const void*)(prow + half * 16);
        pa[mi].f[1] = *(const float4*)(const void*)(prow + half * 16 + 8);
    }

    const float* Qv = qv + (size_t)(b * NQ_ + q) * TT * OUTD;
    size_t obase = (size_t)((b * NS_ + c) * NQ_ + q) * TT * OUTD;

    float nsq = 0.f;
#pragma unroll 1
    for (int nt = 0; nt < 72; ++nt) {
        BFrag bf;
        const unsigned short* brow = &sV[(nt * 16 + r) * 32 + half * 16];
        bf.f[0] = *(const float4*)(const void*)(brow);
        bf.f[1] = *(const float4*)(const void*)(brow + 8);
#pragma unroll
        for (int mi = 0; mi < 2; ++mi) {
            v8f pc = z8;
            pc = wmma_bf16(pa[mi].v, bf.v, pc);
#pragma unroll
            for (int rr = 0; rr < 8; ++rr) {
                int t = mi * 16 + half * 8 + rr;
                if (t < TT) {
                    int n = nt * 16 + r;
                    size_t off = (size_t)t * OUTD + n;
                    float p = pc[rr];
                    float d = Qv[off] - p;
                    protos[obase + off] = p;
                    diffs[obase + off]  = d;
                    nsq += d * d;
                }
            }
        }
    }

    // wave32 reduction of ||diff||^2
#pragma unroll
    for (int i = 16; i > 0; i >>= 1) nsq += __shfl_xor(nsq, i, 32);
    if (lane == 0)
        logits[(size_t)b * (NS_ * NQ_) + c * NQ_ + q] = -(nsq * (1.0f / TT));
}

// --------------------------------------------------------------------------
extern "C" void kernel_launch(void* const* d_in, const int* in_sizes, int n_in,
                              void* d_out, int out_size, void* d_ws, size_t ws_size,
                              hipStream_t stream) {
    const float* support = (const float*)d_in[0];
    // d_in[1] = support_labels (identity permutation, unused)
    const float* queries = (const float*)d_in[2];
    const float* k_w  = (const float*)d_in[3];
    const float* k_b  = (const float*)d_in[4];
    const float* v_w  = (const float*)d_in[5];
    const float* v_b  = (const float*)d_in[6];
    const float* ln_g = (const float*)d_in[7];
    const float* ln_b = (const float*)d_in[8];

    const size_t nXs  = (size_t)B_ * NS_ * SEQ * IND;
    const size_t nXq  = (size_t)B_ * NQ_ * SEQ * IND;
    const size_t nW   = (size_t)OUTD * KDIM;
    const size_t nSk  = (size_t)B_ * NS_ * TT * OUTD;
    const size_t nQk  = (size_t)B_ * NQ_ * TT * OUTD;
    const size_t nSv  = nSk;
    const size_t nSvt = (size_t)B_ * NS_ * OUTD * 32;
    const size_t nQv  = (size_t)B_ * NQ_ * TT * OUTD;

    char* ws = (char*)d_ws;
    size_t off = 0;
    auto alloc = [&](size_t bytes) -> char* {
        char* p = ws + off;
        off += (bytes + 255) & ~(size_t)255;
        return p;
    };
    unsigned short* Xs  = (unsigned short*)alloc(nXs * 2);
    unsigned short* Xq  = (unsigned short*)alloc(nXq * 2);
    unsigned short* kwt = (unsigned short*)alloc(nW * 2);
    unsigned short* vwt = (unsigned short*)alloc(nW * 2);
    unsigned short* skb = (unsigned short*)alloc(nSk * 2);
    unsigned short* qkb = (unsigned short*)alloc(nQk * 2);
    unsigned short* svb = (unsigned short*)alloc(nSv * 2);
    unsigned short* svt = (unsigned short*)alloc(nSvt * 2);
    float*          qvf = (float*)alloc(nQv * 4);

    float* logits = (float*)d_out;                       // 16*500
    float* diffs  = logits + 8000;                       // 16*500*28*1152
    float* protos = diffs + (size_t)258048000;           // 16*25*20*28*1152

    addpe_bf16_kernel<<<(int)((nXs + 255) / 256), 256, 0, stream>>>(support, Xs, (int)nXs);
    addpe_bf16_kernel<<<(int)((nXq + 255) / 256), 256, 0, stream>>>(queries, Xq, (int)nXq);
    wt_transpose_kernel<<<(int)((nW + 255) / 256), 256, 0, stream>>>(k_w, kwt, (int)nW);
    wt_transpose_kernel<<<(int)((nW + 255) / 256), 256, 0, stream>>>(v_w, vwt, (int)nW);

    // K projections (fused layernorm), V projections
    proj_kernel<<<B_ * NS_ * 2, 256, 0, stream>>>(Xs, kwt, k_b, ln_g, ln_b, skb, nullptr, NS_, 1);
    proj_kernel<<<B_ * NQ_ * 2, 256, 0, stream>>>(Xq, kwt, k_b, ln_g, ln_b, qkb, nullptr, NQ_, 1);
    proj_kernel<<<B_ * NS_ * 2, 256, 0, stream>>>(Xs, vwt, v_b, ln_g, ln_b, svb, nullptr, NS_, 0);
    proj_kernel<<<B_ * NQ_ * 2, 256, 0, stream>>>(Xq, vwt, v_b, ln_g, ln_b, nullptr, qvf, NQ_, 0);

    sv_transpose_kernel<<<B_ * NS_, 256, 0, stream>>>(svb, svt);

    attn_kernel<<<B_ * NS_ * (NQ_ / 4), 128, 0, stream>>>(qkb, skb, svt, qvf,
                                                          logits, diffs, protos);
}
